// Emotic_66348654789132
// MI455X (gfx1250) — matrix operand order repaired and checked by reading, not compiled, not run on hardware
//
#include <hip/hip_runtime.h>
#include <hip/hip_bf16.h>
#include <math.h>

typedef __attribute__((ext_vector_type(16))) _Float16 v16h;
typedef __attribute__((ext_vector_type(8)))  float    v8f;
typedef __attribute__((ext_vector_type(4)))  unsigned int u32x4;
typedef __attribute__((ext_vector_type(4)))  int      i32x4;
typedef __attribute__((ext_vector_type(8)))  int      i32x8;

#define LDSP 40  // 32 K-slots + 8 halves pad (bank-conflict avoidance)

// ---------------------------------------------------------------------------
// Fragment loaders following the CDNA5 ISA 16-bit WMMA VGPR layouts:
//   A (16x32): lane L, vgpr v -> m = base + (L&15), k = 2v + (v>=4?8:0) + 8*(L>>4)
//   B (32x16): lane L, vgpr v -> n = base + (L&15), k = 2v + (v>=4?8:0) + 8*(L>>4)
// ---------------------------------------------------------------------------
__device__ __forceinline__ v16h load_a_frag(const _Float16* As, int mbase, int lane) {
  v16h a;
  const int half = lane >> 4;
  const int mrow = mbase + (lane & 15);
#pragma unroll
  for (int v = 0; v < 8; ++v) {
    const int ka = 2 * v + (v >= 4 ? 8 : 0) + half * 8;
    a[2 * v]     = As[mrow * LDSP + ka];
    a[2 * v + 1] = As[mrow * LDSP + ka + 1];
  }
  return a;
}

__device__ __forceinline__ v16h load_b_frag(const _Float16* Bs, int nbase, int lane) {
  v16h b;
  const int half = lane >> 4;
  const int nrow = nbase + (lane & 15);
#pragma unroll
  for (int v = 0; v < 8; ++v) {
    const int kb = 2 * v + (v >= 4 ? 8 : 0) + half * 8;
    b[2 * v]     = Bs[nrow * LDSP + kb];
    b[2 * v + 1] = Bs[nrow * LDSP + kb + 1];
  }
  return b;
}

__device__ __forceinline__ v8f wmma16(v16h a, v16h b, v8f acc) {
  return __builtin_amdgcn_wmma_f32_16x16x32_f16(false, a, false, b, (short)0,
                                                acc, false, false);
}

// ---------------------------------------------------------------------------
// TDM: DMA a 128(row) x 32(col) f16 tile from a row-major [rows, rowStride]
// f16 tensor in global memory into LDS at ldsByteOff, padding each 64B row
// with 16B so the LDS layout matches LDSP=40 halves/row.  D# per ISA §8.3/8.4:
//   group0: count=1, lds_addr, 57-bit global tile addr, type=2
//   group1: data_size=2B, pad_enable (interval code 3 = 16 DW, amount code 3 =
//           4 DW), tensor_dim0/1 = remaining extent (zero-fill OOB),
//           tile_dim0=32, tile_dim1=128, dim0_stride = rowStride elems
// ---------------------------------------------------------------------------
#if __has_builtin(__builtin_amdgcn_tensor_load_to_lds)
#define HAVE_TDM 1
__device__ __forceinline__ void tdm_load_b_tile(const _Float16* gsrc,
                                                long rowStrideElems,
                                                unsigned remRows, unsigned remK,
                                                unsigned ldsByteOff) {
  const unsigned long long ga = (unsigned long long)(size_t)gsrc;
  u32x4 g0;
  g0[0] = 1u;                                       // count=1 (valid), user mode
  g0[1] = ldsByteOff;                               // lds_addr (bytes)
  g0[2] = (unsigned)(ga & 0xFFFFFFFFu);             // global_addr[31:0]
  g0[3] = (unsigned)((ga >> 32) & 0x1FFFFFFu)       // global_addr[56:32]
        | (2u << 30);                               // type=2 ("image")
  i32x8 g1;
  g1[0] = (int)((1u << 16)                          // data_size = 2 bytes
              | (1u << 20)                          // pad_enable
              | (3u << 22)                          // pad_interval: 16 DWORDs
              | (3u << 25));                        // pad_amount: 4 DWORDs
  g1[1] = (int)((remK & 0xFFFFu) << 16);            // tensor_dim0[15:0]
  g1[2] = (int)((remK >> 16) | ((remRows & 0xFFFFu) << 16)); // dim0 hi | dim1 lo
  g1[3] = (int)((remRows >> 16) | (32u << 16));     // dim1 hi | tile_dim0=32
  g1[4] = (int)128u;                                // tile_dim1=128, tile_dim2=0
  const unsigned long long st = (unsigned long long)rowStrideElems;
  g1[5] = (int)(st & 0xFFFFFFFFu);                  // dim0_stride[31:0]
  g1[6] = (int)((st >> 32) & 0xFFFFu);              // dim0_stride[47:32]
  g1[7] = 0;
  const i32x4 gz = {0, 0, 0, 0};
#if __clang_major__ >= 23
  const i32x8 gz8 = {0, 0, 0, 0, 0, 0, 0, 0};
  __builtin_amdgcn_tensor_load_to_lds(g0, g1, gz, gz, gz8, 0);
#else
  __builtin_amdgcn_tensor_load_to_lds(g0, g1, gz, gz, 0);
#endif
}
#else
#define HAVE_TDM 0
#endif

// ---------------------------------------------------------------------------
// Implicit-GEMM conv: C[b, n=oc, m=oh*OW+ow] = relu(sum_k A_im2col * W + bias)
// Block = 256 threads = 8 waves (2M x 4N); block tile 64(M) x 128(N);
// per-wave 2x2 register tile -> 4 v_wmma per K-chunk.  Weights arrive as f16
// (pre-converted once per launch) and are tiled into LDS by the TDM.
// ---------------------------------------------------------------------------
__global__ __launch_bounds__(256) void conv_wmma(
    const float* __restrict__ in, const _Float16* __restrict__ wgt,
    const float* __restrict__ bias, float* __restrict__ out,
    int IC, int H, int W, int rowOff, long bStride, long cStride,
    int OC, int OH, int OW, int KH, int KW, int stride, int pad, int relu) {
  __shared__ _Float16 As[64 * LDSP];
  __shared__ _Float16 Bs[128 * LDSP];
  const int tid  = threadIdx.x;
  const int lane = tid & 31;
  const int wave = tid >> 5;
  const int mt0 = (wave >> 2) * 32;   // wave M offset within block tile
  const int nt0 = (wave & 3) * 32;    // wave N offset within block tile
  const int b  = blockIdx.z;
  const int mBase = blockIdx.y * 64;
  const int nBase = blockIdx.x * 128;
  const int M = OH * OW;
  const int KHW = KH * KW;
  const int K = IC * KHW;
  const float* inb = in + (long)b * bStride;
  const unsigned bsOff = (unsigned)(size_t)(void*)Bs;  // LDS byte offset

  v8f acc00 = {}, acc01 = {}, acc10 = {}, acc11 = {};
  for (int k0 = 0; k0 < K; k0 += 32) {
#pragma unroll
    for (int t = 0; t < 8; ++t) {          // A tile: 64x32 im2col gather
      const int idx = tid + t * 256;
      const int r = idx >> 5, kk = idx & 31;
      const int m = mBase + r, k = k0 + kk;
      float v = 0.f;
      if (m < M && k < K) {
        const int ic = k / KHW, rem = k - ic * KHW;
        const int kh = rem / KW, kw = rem - kh * KW;
        const int oh = m / OW,  ow = m - oh * OW;
        const int ih = oh * stride - pad + kh;
        const int iw = ow * stride - pad + kw;
        if (ih >= 0 && ih < H && iw >= 0 && iw < W)
          v = inb[(long)ic * cStride + (long)(rowOff + ih) * W + iw];
      }
      As[r * LDSP + kk] = (_Float16)v;
    }
#if HAVE_TDM
    if (wave == 0) {                       // one TDM issue per block
      tdm_load_b_tile(wgt + (long)nBase * K + k0, K,
                      (unsigned)(OC - nBase), (unsigned)(K - k0), bsOff);
      __builtin_amdgcn_s_wait_tensorcnt(0);
    }
#else
#pragma unroll
    for (int t = 0; t < 16; ++t) {         // fallback: VALU tile of f16 weights
      const int idx = tid + t * 256;
      const int r = idx >> 5, kk = idx & 31;
      const int n = nBase + r, k = k0 + kk;
      Bs[r * LDSP + kk] = (n < OC && k < K) ? wgt[(long)n * K + k] : (_Float16)0.f;
    }
#endif
    __syncthreads();
    const v16h a0 = load_a_frag(As, mt0, lane);
    const v16h a1 = load_a_frag(As, mt0 + 16, lane);
    const v16h b0 = load_b_frag(Bs, nt0, lane);
    const v16h b1 = load_b_frag(Bs, nt0 + 16, lane);
    acc00 = wmma16(a0, b0, acc00);
    acc01 = wmma16(a0, b1, acc01);
    acc10 = wmma16(a1, b0, acc10);
    acc11 = wmma16(a1, b1, acc11);
    __syncthreads();
  }
  // epilogue: 2x2 tiles, C/D layout M = r + 8*(lane>=16), N = lane&15
#pragma unroll
  for (int ti = 0; ti < 2; ++ti) {
#pragma unroll
    for (int tj = 0; tj < 2; ++tj) {
      const v8f acc = (ti == 0) ? (tj == 0 ? acc00 : acc01)
                                : (tj == 0 ? acc10 : acc11);
      const int nc = nBase + nt0 + tj * 16 + (lane & 15);
#pragma unroll
      for (int r = 0; r < 8; ++r) {
        const int m = mBase + mt0 + ti * 16 + r + ((lane >> 4) << 3);
        if (m < M && nc < OC) {
          float v = acc[r] + bias[nc];
          if (relu) v = fmaxf(v, 0.f);
          out[((long)b * OC + nc) * (long)M + m] = v;
        }
      }
    }
  }
}

// ---------------------------------------------------------------------------
// Dense GEMM: C[m, coff+n] = act(A[m,:K] . Wt[n,:K] + bias[n]); A lda-strided.
// Kept on the single-pass f32-read + VALU-convert path: the big FC layers are
// HBM-bound (one streaming read of the f32 weights is the bandwidth optimum).
// ---------------------------------------------------------------------------
__global__ __launch_bounds__(256) void gemm_wmma(
    const float* __restrict__ A, int lda,
    const float* __restrict__ Wt, const float* __restrict__ bias,
    float* __restrict__ C, int ldc, int coff,
    int M, int N, int K, int relu) {
  __shared__ _Float16 As[64 * LDSP];
  __shared__ _Float16 Bs[128 * LDSP];
  const int tid  = threadIdx.x;
  const int lane = tid & 31;
  const int wave = tid >> 5;
  const int mt0 = (wave >> 2) * 32;
  const int nt0 = (wave & 3) * 32;
  const int mBase = blockIdx.y * 64;
  const int nBase = blockIdx.x * 128;

  v8f acc00 = {}, acc01 = {}, acc10 = {}, acc11 = {};
  for (int k0 = 0; k0 < K; k0 += 32) {
#pragma unroll
    for (int t = 0; t < 8; ++t) {
      const int idx = tid + t * 256;
      const int r = idx >> 5, kk = idx & 31;
      const int m = mBase + r, k = k0 + kk;
      As[r * LDSP + kk] = (_Float16)((m < M && k < K) ? A[(long)m * lda + k] : 0.f);
    }
    {   // prefetch next weight K-chunk (global_prefetch_b8)
      const int pn = nBase + (tid >> 1);
      if (k0 + 32 < K && pn < N)
        __builtin_prefetch(&Wt[(long)pn * K + k0 + 32], 0, 1);
    }
#pragma unroll
    for (int t = 0; t < 16; ++t) {
      const int idx = tid + t * 256;
      const int r = idx >> 5, kk = idx & 31;
      const int n = nBase + r, k = k0 + kk;
      Bs[r * LDSP + kk] = (_Float16)((n < N && k < K) ? Wt[(long)n * K + k] : 0.f);
    }
    __syncthreads();
    const v16h a0 = load_a_frag(As, mt0, lane);
    const v16h a1 = load_a_frag(As, mt0 + 16, lane);
    const v16h b0 = load_b_frag(Bs, nt0, lane);
    const v16h b1 = load_b_frag(Bs, nt0 + 16, lane);
    acc00 = wmma16(a0, b0, acc00);
    acc01 = wmma16(a0, b1, acc01);
    acc10 = wmma16(a1, b0, acc10);
    acc11 = wmma16(a1, b1, acc11);
    __syncthreads();
  }
#pragma unroll
  for (int ti = 0; ti < 2; ++ti) {
#pragma unroll
    for (int tj = 0; tj < 2; ++tj) {
      const v8f acc = (ti == 0) ? (tj == 0 ? acc00 : acc01)
                                : (tj == 0 ? acc10 : acc11);
      const int n = nBase + nt0 + tj * 16 + (lane & 15);
#pragma unroll
      for (int r = 0; r < 8; ++r) {
        const int m = mBase + mt0 + ti * 16 + r + ((lane >> 4) << 3);
        if (m < M && n < N) {
          float v = acc[r] + bias[n];
          if (relu) v = fmaxf(v, 0.f);
          C[(long)m * ldc + coff + n] = v;
        }
      }
    }
  }
}

// ------------------------- weight f32 -> f16 convert -----------------------
__global__ void f32_to_f16(const float* __restrict__ src, _Float16* __restrict__ dst,
                           int n) {
  const int i = blockIdx.x * 256 + threadIdx.x;
  if (i < n) dst[i] = (_Float16)src[i];
}

// ------------------------------ pooling ------------------------------------
__global__ void maxpool3s2(const float* __restrict__ in, float* __restrict__ out,
                           int B, int C, int H, int W, int OH, int OW) {
  const int idx = blockIdx.x * 256 + threadIdx.x;
  const int total = B * C * OH * OW;
  if (idx >= total) return;
  int ow = idx % OW, t = idx / OW;
  int oh = t % OH;  t /= OH;
  int c  = t % C;   const int b = t / C;
  const float* p = in + ((long)b * C + c) * (long)H * W;
  float m = -3.402823466e38f;
#pragma unroll
  for (int i = 0; i < 3; ++i)
#pragma unroll
    for (int j = 0; j < 3; ++j)
      m = fmaxf(m, p[(oh * 2 + i) * W + (ow * 2 + j)]);
  out[idx] = m;
}

__global__ void avgpool2s1(const float* __restrict__ in, float* __restrict__ out,
                           int B, int C, int H, int W, int OH, int OW) {
  const int idx = blockIdx.x * 256 + threadIdx.x;
  const int total = B * C * OH * OW;
  if (idx >= total) return;
  int ow = idx % OW, t = idx / OW;
  int oh = t % OH;  t /= OH;
  int c  = t % C;   const int b = t / C;
  const float* p = in + ((long)b * C + c) * (long)H * W;
  out[idx] = 0.25f * (p[oh * W + ow] + p[oh * W + ow + 1] +
                      p[(oh + 1) * W + ow] + p[(oh + 1) * W + ow + 1]);
}

// ----------------------- semantic color-hit mask ---------------------------
__global__ void zero_ints(int* p, int n) {
  const int i = blockIdx.x * blockDim.x + threadIdx.x;
  if (i < n) p[i] = 0;
}

__global__ void color_hit(const float* __restrict__ x, const int* __restrict__ colors,
                          int* __restrict__ hit) {
  __shared__ int col[450];
  for (int i = threadIdx.x; i < 450; i += 256) col[i] = colors[i];
  __syncthreads();
  const int idx = blockIdx.x * 256 + threadIdx.x;
  if (idx >= 32 * 256 * 256) return;
  const int w = idx & 255;
  int t = idx >> 8;
  const int h = t & 255;
  const int b = t >> 8;
  const float* base = x + (long)b * 3 * 768 * 256;
  const float r = base[(0 * 768 + 512 + h) * 256 + w];
  const float g = base[(1 * 768 + 512 + h) * 256 + w];
  const float bb = base[(2 * 768 + 512 + h) * 256 + w];
  for (int c = 0; c < 150; ++c)
    if (r == (float)col[c * 3] && g == (float)col[c * 3 + 1] && bb == (float)col[c * 3 + 2])
      atomicOr(&hit[c], 1);
}

// ----------------------- BiLSTM (T=32 over batch axis) ---------------------
__device__ __forceinline__ float sigm(float x) { return 1.f / (1.f + expf(-x)); }

__global__ void lstm2_kernel(const int* __restrict__ hit,
    const float* __restrict__ wih0, const float* __restrict__ whh0,
    const float* __restrict__ bih0, const float* __restrict__ bhh0,
    const float* __restrict__ wih1, const float* __restrict__ whh1,
    const float* __restrict__ bih1, const float* __restrict__ bhh1,
    float* __restrict__ fsbuf) {
  const int n = blockIdx.x;            // 0..149
  const int tid = threadIdx.x;         // 0..31
  const int dir = tid >> 4, j = tid & 15;
  __shared__ float h0[32][32];
  __shared__ float h1[32][32];
  __shared__ float hc[32];
  const float xv = hit[n] ? 1.0f : 0.0f;

  { // layer 0 (input dim 1)
    const float* wih = wih0 + dir * 64;
    const float* whh = whh0 + dir * 64 * 16;
    const float* bi  = bih0 + dir * 64;
    const float* bh  = bhh0 + dir * 64;
    float cs = 0.f;
    hc[tid] = 0.f;
    __syncthreads();
    for (int s = 0; s < 32; ++s) {
      const int t = dir ? (31 - s) : s;
      float gi = wih[j] * xv + bi[j] + bh[j];
      float gf = wih[16 + j] * xv + bi[16 + j] + bh[16 + j];
      float gg = wih[32 + j] * xv + bi[32 + j] + bh[32 + j];
      float go = wih[48 + j] * xv + bi[48 + j] + bh[48 + j];
      for (int u = 0; u < 16; ++u) {
        const float hu = hc[dir * 16 + u];
        gi += whh[j * 16 + u] * hu;        gf += whh[(16 + j) * 16 + u] * hu;
        gg += whh[(32 + j) * 16 + u] * hu; go += whh[(48 + j) * 16 + u] * hu;
      }
      __syncthreads();
      cs = sigm(gf) * cs + sigm(gi) * tanhf(gg);
      const float hn = sigm(go) * tanhf(cs);
      hc[tid] = hn;
      h0[t][tid] = hn;
      __syncthreads();
    }
  }
  { // layer 1 (input dim 32 = concat of layer-0 dirs)
    const float* wih = wih1 + dir * 64 * 32;
    const float* whh = whh1 + dir * 64 * 16;
    const float* bi  = bih1 + dir * 64;
    const float* bh  = bhh1 + dir * 64;
    float cs = 0.f;
    __syncthreads();
    hc[tid] = 0.f;
    __syncthreads();
    for (int s = 0; s < 32; ++s) {
      const int t = dir ? (31 - s) : s;
      float gi = bi[j] + bh[j];
      float gf = bi[16 + j] + bh[16 + j];
      float gg = bi[32 + j] + bh[32 + j];
      float go = bi[48 + j] + bh[48 + j];
      for (int k = 0; k < 32; ++k) {
        const float xk = h0[t][k];
        gi += wih[j * 32 + k] * xk;        gf += wih[(16 + j) * 32 + k] * xk;
        gg += wih[(32 + j) * 32 + k] * xk; go += wih[(48 + j) * 32 + k] * xk;
      }
      for (int u = 0; u < 16; ++u) {
        const float hu = hc[dir * 16 + u];
        gi += whh[j * 16 + u] * hu;        gf += whh[(16 + j) * 16 + u] * hu;
        gg += whh[(32 + j) * 16 + u] * hu; go += whh[(48 + j) * 16 + u] * hu;
      }
      __syncthreads();
      cs = sigm(gf) * cs + sigm(gi) * tanhf(gg);
      const float hn = sigm(go) * tanhf(cs);
      hc[tid] = hn;
      h1[t][tid] = hn;
      __syncthreads();
    }
  }
  { // fs[t][n] = mean over concat hidden dims
    const int t = tid;
    float sum = 0.f;
    for (int k = 0; k < 32; ++k) sum += h1[t][k];
    fsbuf[t * 150 + n] = sum * (1.0f / 32.0f);
  }
}

// ----------------------------- host orchestration --------------------------
static void run_alexnet(const float* x, int rowOff, void* const* P,
                        float* buf0, float* buf1, float* fuse_in, _Float16* wgt_h,
                        int ldc, int coff, int ncls, hipStream_t s) {
  const float *c1w = (const float*)P[0],  *c1b = (const float*)P[1];
  const float *c2w = (const float*)P[2],  *c2b = (const float*)P[3];
  const float *c3w = (const float*)P[4],  *c3b = (const float*)P[5];
  const float *c4w = (const float*)P[6],  *c4b = (const float*)P[7];
  const float *c5w = (const float*)P[8],  *c5b = (const float*)P[9];
  const float *f1w = (const float*)P[10], *f1b = (const float*)P[11];
  const float *f2w = (const float*)P[12], *f2b = (const float*)P[13];
  const float *f3w = (const float*)P[14], *f3b = (const float*)P[15];
  dim3 blk(256);
  auto cg = [](int M, int N) {
    return dim3((unsigned)((N + 127) / 128), (unsigned)((M + 63) / 64), 32u);
  };
  auto cvt = [&](const float* w, int n) {
    f32_to_f16<<<(n + 255) / 256, 256, 0, s>>>(w, wgt_h, n);
  };
  cvt(c1w, 64 * 363);
  conv_wmma<<<cg(63 * 63, 64), blk, 0, s>>>(x, wgt_h, c1b, buf0, 3, 256, 256, rowOff,
      3L * 768 * 256, 768L * 256, 64, 63, 63, 11, 11, 4, 2, 1);
  maxpool3s2<<<(32 * 64 * 31 * 31 + 255) / 256, blk, 0, s>>>(buf0, buf1, 32, 64, 63, 63, 31, 31);
  cvt(c2w, 192 * 1600);
  conv_wmma<<<cg(31 * 31, 192), blk, 0, s>>>(buf1, wgt_h, c2b, buf0, 64, 31, 31, 0,
      64L * 31 * 31, 31L * 31, 192, 31, 31, 5, 5, 1, 2, 1);
  maxpool3s2<<<(32 * 192 * 15 * 15 + 255) / 256, blk, 0, s>>>(buf0, buf1, 32, 192, 31, 31, 15, 15);
  cvt(c3w, 384 * 1728);
  conv_wmma<<<cg(225, 384), blk, 0, s>>>(buf1, wgt_h, c3b, buf0, 192, 15, 15, 0,
      192L * 225, 225, 384, 15, 15, 3, 3, 1, 1, 1);
  cvt(c4w, 256 * 3456);
  conv_wmma<<<cg(225, 256), blk, 0, s>>>(buf0, wgt_h, c4b, buf1, 384, 15, 15, 0,
      384L * 225, 225, 256, 15, 15, 3, 3, 1, 1, 1);
  cvt(c5w, 256 * 2304);
  conv_wmma<<<cg(225, 256), blk, 0, s>>>(buf1, wgt_h, c5b, buf0, 256, 15, 15, 0,
      256L * 225, 225, 256, 15, 15, 3, 3, 1, 1, 1);
  maxpool3s2<<<(32 * 256 * 7 * 7 + 255) / 256, blk, 0, s>>>(buf0, buf1, 32, 256, 15, 15, 7, 7);
  avgpool2s1<<<(32 * 256 * 6 * 6 + 255) / 256, blk, 0, s>>>(buf1, buf0, 32, 256, 7, 7, 6, 6);
  gemm_wmma<<<dim3(32, 1), blk, 0, s>>>(buf0, 9216, f1w, f1b, buf1, 4096, 0, 32, 4096, 9216, 1);
  gemm_wmma<<<dim3(32, 1), blk, 0, s>>>(buf1, 4096, f2w, f2b, buf0, 4096, 0, 32, 4096, 4096, 1);
  gemm_wmma<<<dim3((unsigned)((ncls + 127) / 128), 1), blk, 0, s>>>(
      buf0, 4096, f3w, f3b, fuse_in, ldc, coff, 32, ncls, 4096, 0);
}

extern "C" void kernel_launch(void* const* d_in, const int* in_sizes, int n_in,
                              void* d_out, int out_size, void* d_ws, size_t ws_size,
                              hipStream_t stream) {
  // d_in layout (setup_inputs flattened): 0:x 1:colors 2..17:ctx 18..33:body
  // 34..41:lstm(wih0,whh0,bih0,bhh0,wih1,whh1,bih1,bhh1) 42:fc1_w 43:fc1_b
  // 44:lift_w 45:lift_b
  const float* X      = (const float*)d_in[0];
  const int*   colors = (const int*)d_in[1];
  void* const* ctxP   = d_in + 2;
  void* const* bodP   = d_in + 18;
  const float* wih0 = (const float*)d_in[34];
  const float* whh0 = (const float*)d_in[35];
  const float* bih0 = (const float*)d_in[36];
  const float* bhh0 = (const float*)d_in[37];
  const float* wih1 = (const float*)d_in[38];
  const float* whh1 = (const float*)d_in[39];
  const float* bih1 = (const float*)d_in[40];
  const float* bhh1 = (const float*)d_in[41];
  const float* fc1w = (const float*)d_in[42];
  const float* fc1b = (const float*)d_in[43];
  const float* liftw = (const float*)d_in[44];
  const float* liftb = (const float*)d_in[45];
  float* out = (float*)d_out;  // [32*512 fuse | 32*64 feat_sem]

  // workspace: two ping-pong activation buffers (max 8,128,512 floats each),
  // fuse input [32,1365], fs [32,150], hit[150], f16 conv-weight staging
  // (max 884,736 halves) -> ~68 MB total
  float*     buf0    = (float*)d_ws;
  float*     buf1    = buf0 + 8200000;
  float*     fuse_in = buf1 + 8200000;
  float*     fsbuf   = fuse_in + 32 * 1365;
  int*       hit     = (int*)(fsbuf + 32 * 150);
  _Float16*  wgt_h   = (_Float16*)(hit + 160);

  // semantic color-hit vector
  zero_ints<<<1, 256, 0, stream>>>(hit, 150);
  color_hit<<<(32 * 256 * 256 + 255) / 256, 256, 0, stream>>>(X, colors, hit);

  // two AlexNets -> fuse_in[:, 0:365] and fuse_in[:, 365:1365]
  run_alexnet(X, 0,   ctxP, buf0, buf1, fuse_in, wgt_h, 1365, 0,   365,  stream);
  run_alexnet(X, 256, bodP, buf0, buf1, fuse_in, wgt_h, 1365, 365, 1000, stream);

  // fuse: [32,1365] @ fc1_w^T + fc1_b -> out[0:16384]
  gemm_wmma<<<dim3(4, 1), 256, 0, stream>>>(fuse_in, 1365, fc1w, fc1b,
                                            out, 512, 0, 32, 512, 1365, 0);

  // BiLSTM over batch-as-time, mean over hidden dims -> fs[32,150]
  lstm2_kernel<<<150, 32, 0, stream>>>(hit, wih0, whh0, bih0, bhh0,
                                       wih1, whh1, bih1, bhh1, fsbuf);
  // lift: [32,150] @ lift_w^T + lift_b -> out[16384:18432]
  gemm_wmma<<<dim3(1, 1), 256, 0, stream>>>(fsbuf, 150, liftw, liftb,
                                            out + 32 * 512, 64, 0, 32, 64, 150, 0);
}
